// TransformerWithCnnModel_82497731822094
// MI455X (gfx1250) — matrix operand order repaired and checked by reading, not compiled, not run on hardware
//
#include <hip/hip_runtime.h>
#include <hip/hip_bf16.h>

typedef __attribute__((ext_vector_type(16))) _Float16     v16h;
typedef __attribute__((ext_vector_type(8)))  float        v8f;
typedef __attribute__((ext_vector_type(4)))  unsigned int v4u;
typedef __attribute__((ext_vector_type(8)))  int          v8i;
typedef __attribute__((ext_vector_type(4)))  int          v4i;

static __device__ __forceinline__ v8f wmma16(v16h a, v16h b, v8f c) {
  // D = A(16x32 f16) * B(32x16 f16) + C(16x16 f32)
  return __builtin_amdgcn_wmma_f32_16x16x32_f16(false, a, false, b, (short)0, c,
                                                false, false);
}

// 16-bit A-matrix 16x32 lane layout (contiguous K, matching B):
// lane<16: M=lane, K in {0..15}; lane>=16: M=lane-16, K in {16..31}
static __device__ __forceinline__ int a_kidx(int lane, int e) {
  return (lane < 16) ? e : (e + 16);
}
// 16-bit B-matrix 32x16: lane<16 -> K=e, lane>=16 -> K=e+16 (contiguous in e!)
static __device__ __forceinline__ int b_koff(int lane) {
  return (lane < 16) ? 0 : 16;
}

// ---------------------------------------------------------------------------
// Tensor Data Mover: 2D tile (tile1 rows x tile0 elems of 2 bytes) -> LDS.
// D# packing per CDNA5 ISA.
// ---------------------------------------------------------------------------
#if __has_builtin(__builtin_amdgcn_tensor_load_to_lds)
#define HAVE_TDM 1
#else
#define HAVE_TDM 0
#endif

#if HAVE_TDM
static __device__ __forceinline__ void tdm_load_tile_f16(
    void* ldsDst, const _Float16* gsrc,
    unsigned dim0, unsigned dim1, unsigned tile0, unsigned tile1,
    unsigned stride0) {
  unsigned long long ga = (unsigned long long)(uintptr_t)gsrc;
  unsigned ldsOff = (unsigned)(uintptr_t)ldsDst;   // generic LDS addr[31:0] == LDS offset
  v4u g0;
  g0.x = 1u;                                        // count=1, user descriptor
  g0.y = ldsOff;                                    // lds_addr (bits 63:32)
  g0.z = (unsigned)ga;                              // global_addr lo
  g0.w = (unsigned)((ga >> 32) & 0x01FFFFFFu) | (2u << 30);  // addr hi | type=2
  v8i g1;
  g1[0] = (int)(1u << 16);                          // data_size=1 (2 bytes)
  g1[1] = (int)((dim0 & 0xFFFFu) << 16);            // tensor_dim0 [79:48] lo16
  g1[2] = (int)(((dim0 >> 16) & 0xFFFFu) | ((dim1 & 0xFFFFu) << 16));  // dim0 hi / dim1 lo
  g1[3] = (int)(((dim1 >> 16) & 0xFFFFu) | ((tile0 & 0xFFFFu) << 16)); // dim1 hi / tile_dim0
  g1[4] = (int)(tile1 & 0xFFFFu);                   // tile_dim1 (tile_dim2=0)
  g1[5] = (int)stride0;                             // tensor_dim0_stride [191:160]
  g1[6] = 0;
  g1[7] = 0;
  v4i gz = {0, 0, 0, 0};
#if __clang_major__ >= 23
  __builtin_amdgcn_tensor_load_to_lds(g0, g1, gz, gz, (v8i)0, 0);
#else
  __builtin_amdgcn_tensor_load_to_lds(g0, g1, gz, gz, 0);
#endif
}
#endif

// ---------------------------------------------------------------------------
// Prep: f16 weight repack (zero-padded tiles), BN folding, concat scaffolding
// ---------------------------------------------------------------------------
__global__ __launch_bounds__(256)
void k_prep(const float* __restrict__ qv,
            const float* __restrict__ c1w, const float* __restrict__ c2w,
            const float* __restrict__ f1w,
            const float* __restrict__ g1, const float* __restrict__ b1,
            const float* __restrict__ m1, const float* __restrict__ v1,
            const float* __restrict__ g2, const float* __restrict__ b2,
            const float* __restrict__ m2, const float* __restrict__ v2,
            _Float16* __restrict__ w1h, _Float16* __restrict__ w2h,
            _Float16* __restrict__ f1wh,
            _Float16* __restrict__ fcin, _Float16* __restrict__ fcact,
            float* __restrict__ bnp) {
  int g = blockIdx.x * blockDim.x + threadIdx.x;
  int stride = gridDim.x * blockDim.x;
  for (int i = g; i < 128 * 800; i += stride) {
    int o = i / 800, k = i % 800;
    w1h[i] = (o < 120) ? (_Float16)c1w[o * 800 + k] : (_Float16)0.0f;
  }
  for (int i = g; i < 128 * 608; i += stride) {
    int o = i / 608, k = i % 608;
    w2h[i] = (o < 120 && k < 600) ? (_Float16)c2w[o * 600 + k] : (_Float16)0.0f;
  }
  for (int i = g; i < 512 * 1536; i += stride) {
    int n = i / 1536, k = i % 1536;
    f1wh[i] = (n < 500 && k < 1504) ? (_Float16)f1w[n * 1504 + k] : (_Float16)0.0f;
  }
  for (int i = g; i < 256 * 1536; i += stride) {
    int bb = i / 1536, k = i % 1536;
    fcin[i] = (k < 64) ? (_Float16)qv[bb * 64 + k] : (_Float16)0.0f;
  }
  for (int i = g; i < 256 * 576; i += stride) {
    int bb = i / 576, k = i % 576;
    fcact[i] = (k < 64) ? (_Float16)qv[bb * 64 + k] : (_Float16)0.0f;
  }
  for (int i = g; i < 120; i += stride) {
    float s1 = g1[i] * rsqrtf(v1[i] + 1e-5f);
    bnp[i]       = s1;
    bnp[120 + i] = b1[i] - m1[i] * s1;
    float s2 = g2[i] * rsqrtf(v2[i] + 1e-5f);
    bnp[240 + i] = s2;
    bnp[360 + i] = b2[i] - m2[i] * s2;
  }
}

// ---------------------------------------------------------------------------
// Fused 8x8 DCT (WMMA GEMM M=1024 patches, N=64 coeffs, K=64) + hard histogram.
// gamma=1e6 sigmoid differences == exact bin counts: bin = floor(x+80) in [0,160).
// Branchless binning: invalid lanes (DC coeff / out-of-range) hit a dummy slot,
// so every lane issues exactly one unconditional ds_add_u32 (EXEC stays all-1s).
// ---------------------------------------------------------------------------
#define HIST_DUMMY (63 * 160)

__global__ __launch_bounds__(256)
void k_dct_hist(const float* __restrict__ x, const float* __restrict__ basis,
                _Float16* __restrict__ featH) {
  alignas(32) __shared__ _Float16 basisH[64 * 64];    // [n][k]
  alignas(32) __shared__ _Float16 patchB[128 * 64];   // [localPatch][k]
  __shared__ unsigned int hist[63 * 160 + 1];         // [coeff-1][bin] + dummy
  const int tid = threadIdx.x;
  const int lane = tid & 31, wave = tid >> 5;
  const int blk = blockIdx.x;

  for (int i = tid; i < 64 * 64; i += 256) basisH[i] = (_Float16)basis[i];
  for (int i = tid; i < 63 * 160 + 1; i += 256) hist[i] = 0u;
  __syncthreads();

  const float* img = x + (size_t)blk * 65536;
  const int m = lane & 15;
  const int boff = b_koff(lane);

  for (int it = 0; it < 8; ++it) {            // 8 x 128 patches = 1024
    const int pbase = it * 128;
    for (int i = tid; i < 128 * 64; i += 256) {
      int lp = i >> 6, k = i & 63;
      int patch = pbase + lp;
      int py = patch >> 5, px = patch & 31;
      int ii = k >> 3, jj = k & 7;
      patchB[i] = (_Float16)img[(py * 8 + ii) * 256 + px * 8 + jj];
    }
    __syncthreads();

    v16h a0, a1;
#pragma unroll
    for (int e = 0; e < 16; ++e) {
      int base = (wave * 16 + m) * 64;
      a0[e] = patchB[base + a_kidx(lane, e)];
      a1[e] = patchB[base + 32 + a_kidx(lane, e)];
    }
#pragma unroll
    for (int nt = 0; nt < 4; ++nt) {
      int n = nt * 16 + m;
      const v16h bv0 = *(const v16h*)(basisH + n * 64 + boff);
      const v16h bv1 = *(const v16h*)(basisH + n * 64 + 32 + boff);
      v8f acc = {};
      acc = wmma16(a0, bv0, acc);
      acc = wmma16(a1, bv1, acc);
      int coeff = n;                          // N index = coefficient id
#pragma unroll
      for (int r = 0; r < 8; ++r) {
        int t = (int)floorf(acc[r] + 80.0f);
        bool valid = (coeff >= 1) & ((unsigned)t < 160u);
        int idx = valid ? ((coeff - 1) * 160 + t) : HIST_DUMMY;
        atomicAdd(&hist[idx], 1u);            // unconditional, branch-free
      }
    }
    __syncthreads();
  }

  _Float16* frow = featH + (size_t)blk * (160 * 64);
  for (int i = tid; i < 160 * 64; i += 256) {
    int t = i >> 6, p = i & 63;
    float v = (p < 63) ? (float)hist[p * 160 + t] * (1.0f / 1024.0f) : 0.0f;
    frow[i] = (_Float16)v;
  }
}

// ---------------------------------------------------------------------------
// conv1 (im2col GEMM M=120,N=59,K=800) + bias->relu->BN->relu->maxpool2.
// Weight tiles staged by the Tensor Data Mover; im2col staged N-major.
// ---------------------------------------------------------------------------
__global__ __launch_bounds__(256)
void k_conv1(const _Float16* __restrict__ featH, const _Float16* __restrict__ w1h,
             const float* __restrict__ cb1, const float* __restrict__ s1,
             const float* __restrict__ sh1, _Float16* __restrict__ out1h) {
  alignas(32) __shared__ _Float16 As[128 * 32];   // [o][k]
  alignas(32) __shared__ _Float16 Bs[64 * 32];    // [pos][k]  (N-major)
  const int tid = threadIdx.x, lane = tid & 31, wave = tid >> 5;
  const int blk = blockIdx.x;
  const _Float16* fr = featH + (size_t)blk * (160 * 64);
  const int m = lane & 15;
  const int boff = b_koff(lane);

  v8f acc[4] = {};
  for (int kc = 0; kc < 25; ++kc) {           // K = 800 = 25 x 32
    const int kbase = kc * 32;
#if HAVE_TDM
    if (wave == 0)                            // one DMA for the 128x32 tile
      tdm_load_tile_f16(As, w1h + kbase, 800u, 128u, 32u, 128u, 800u);
#else
    for (int i = tid; i < 2048; i += 256) {
      int r = i >> 4, c = i & 15;
      ((unsigned int*)As)[i] = ((const unsigned int*)(w1h + r * 800 + kbase))[c];
    }
#endif
    for (int i = tid; i < 2048; i += 256) {   // im2col chunk, N-major [l][kr]
      int l = i >> 5, kr = i & 31;
      int k = kbase + kr;
      int t = k / 5, kk = k % 5;
      _Float16 v = (_Float16)0.0f;
      if (l + kk < 63) v = fr[t * 64 + l + kk];
      Bs[i] = v;
    }
#if HAVE_TDM
    if (wave == 0) __builtin_amdgcn_s_wait_tensorcnt(0);
#endif
    __syncthreads();
    v16h av;
#pragma unroll
    for (int e = 0; e < 16; ++e) av[e] = As[(wave * 16 + m) * 32 + a_kidx(lane, e)];
#pragma unroll
    for (int nt = 0; nt < 4; ++nt) {
      const v16h bv = *(const v16h*)(Bs + (nt * 16 + m) * 32 + boff);
      acc[nt] = wmma16(av, bv, acc[nt]);
    }
    __syncthreads();
  }
  // fused epilogue: bias, relu, BN, relu, pairwise maxpool via shfl_xor
#pragma unroll
  for (int nt = 0; nt < 4; ++nt) {
    int l = nt * 16 + m;
#pragma unroll
    for (int r = 0; r < 8; ++r) {
      int o = wave * 16 + r + ((lane < 16) ? 0 : 8);
      int oc = (o < 120) ? o : 119;           // clamp -> unconditional loads
      float y = fmaxf(acc[nt][r] + cb1[oc], 0.0f);
      y = fmaxf(y * s1[oc] + sh1[oc], 0.0f);
      float p = __shfl_xor(y, 1, 32);
      float mx = fmaxf(y, p);
      if (o < 120 && (l & 1) == 0 && l < 58)  // 59 -> 29 pooled positions
        out1h[(size_t)blk * (120 * 32) + o * 32 + (l >> 1)] = (_Float16)mx;
    }
  }
}

// ---------------------------------------------------------------------------
// conv2 (GEMM M=120,N=25,K=600) + bias->relu->BN->relu->maxpool2.
// Pooled result goes straight into the fc1 concat buffer fcin[b][64 + o*12 + lp].
// ---------------------------------------------------------------------------
__global__ __launch_bounds__(256)
void k_conv2(const _Float16* __restrict__ out1h, const _Float16* __restrict__ w2h,
             const float* __restrict__ cb2, const float* __restrict__ s2,
             const float* __restrict__ sh2, _Float16* __restrict__ fcin) {
  alignas(32) __shared__ _Float16 As[128 * 32];
  alignas(32) __shared__ _Float16 Bs[32 * 32];    // [pos][k] N-major
  const int tid = threadIdx.x, lane = tid & 31, wave = tid >> 5;
  const int blk = blockIdx.x;
  const _Float16* ir = out1h + (size_t)blk * (120 * 32);
  const int m = lane & 15;
  const int boff = b_koff(lane);

  v8f acc[2] = {};
  for (int kc = 0; kc < 19; ++kc) {           // K = 608 (600 + pad) = 19 x 32
    const int kbase = kc * 32;
#if HAVE_TDM
    if (wave == 0)
      tdm_load_tile_f16(As, w2h + kbase, 608u, 128u, 32u, 128u, 608u);
#else
    for (int i = tid; i < 2048; i += 256) {
      int r = i >> 4, c = i & 15;
      ((unsigned int*)As)[i] = ((const unsigned int*)(w2h + r * 608 + kbase))[c];
    }
#endif
    for (int i = tid; i < 1024; i += 256) {   // im2col N-major [l][kr]
      int l = i >> 5, kr = i & 31;
      int k = kbase + kr;
      int t = k / 5, kk = k % 5;
      _Float16 v = (_Float16)0.0f;
      if (k < 600 && l + kk < 29) v = ir[t * 32 + l + kk];
      Bs[i] = v;
    }
#if HAVE_TDM
    if (wave == 0) __builtin_amdgcn_s_wait_tensorcnt(0);
#endif
    __syncthreads();
    v16h av;
#pragma unroll
    for (int e = 0; e < 16; ++e) av[e] = As[(wave * 16 + m) * 32 + a_kidx(lane, e)];
#pragma unroll
    for (int nt = 0; nt < 2; ++nt) {
      const v16h bv = *(const v16h*)(Bs + (nt * 16 + m) * 32 + boff);
      acc[nt] = wmma16(av, bv, acc[nt]);
    }
    __syncthreads();
  }
#pragma unroll
  for (int nt = 0; nt < 2; ++nt) {
    int l = nt * 16 + m;
#pragma unroll
    for (int r = 0; r < 8; ++r) {
      int o = wave * 16 + r + ((lane < 16) ? 0 : 8);
      int oc = (o < 120) ? o : 119;
      float y = fmaxf(acc[nt][r] + cb2[oc], 0.0f);
      y = fmaxf(y * s2[oc] + sh2[oc], 0.0f);
      float p = __shfl_xor(y, 1, 32);
      float mx = fmaxf(y, p);
      if (o < 120 && (l & 1) == 0 && l < 24)  // 25 -> 12 pooled positions
        fcin[(size_t)blk * 1536 + 64 + o * 12 + (l >> 1)] = (_Float16)mx;
    }
  }
}

// ---------------------------------------------------------------------------
// fc1: GEMM M=256(batch), N=500(pad 512), K=1504(pad 1536) + relu.
// Activations tile via TDM; weight fragments are contiguous 32B global loads.
// ---------------------------------------------------------------------------
__global__ __launch_bounds__(256)
void k_fc1(const _Float16* __restrict__ fcin, const _Float16* __restrict__ f1wh,
           const float* __restrict__ f1b, _Float16* __restrict__ fcact) {
  alignas(32) __shared__ _Float16 As[16 * 32];
  const int tid = threadIdx.x, lane = tid & 31, wave = tid >> 5;
  const int mt = blockIdx.x >> 2, ng = blockIdx.x & 3;
  const int nt = ng * 8 + wave;
  const int m = lane & 15;
  const int n = nt * 16 + m;

  v8f acc = {};
  for (int kc = 0; kc < 48; ++kc) {           // K = 1536 = 48 x 32
    const int kbase = kc * 32;
#if HAVE_TDM
    if (wave == 0) {
      tdm_load_tile_f16(As, fcin + (size_t)(mt * 16) * 1536 + kbase,
                        1536u, 16u, 32u, 16u, 1536u);
      __builtin_amdgcn_s_wait_tensorcnt(0);
    }
#else
    {
      int r = tid >> 4, c = tid & 15;
      ((unsigned int*)As)[tid] =
          ((const unsigned int*)(fcin + (mt * 16 + r) * 1536 + kbase))[c];
    }
#endif
    __syncthreads();
    v16h av;
#pragma unroll
    for (int e = 0; e < 16; ++e) av[e] = As[m * 32 + a_kidx(lane, e)];
    const v16h bv = *reinterpret_cast<const v16h*>(
        f1wh + (size_t)n * 1536 + kbase + ((lane < 16) ? 0 : 16));
    acc = wmma16(av, bv, acc);
    __syncthreads();
  }
  float bias = f1b[(n < 500) ? n : 499];
#pragma unroll
  for (int r = 0; r < 8; ++r) {
    int bRow = mt * 16 + r + ((lane < 16) ? 0 : 8);
    float y = fmaxf(acc[r] + bias, 0.0f);
    if (n < 500) fcact[(size_t)bRow * 576 + 64 + n] = (_Float16)y;
  }
}

// ---------------------------------------------------------------------------
// fc2: out[b][j] = fcact[b][:564] . fc2_w[j] + fc2_b[j]  (tiny, VALU)
// ---------------------------------------------------------------------------
__global__ __launch_bounds__(256)
void k_fc2(const _Float16* __restrict__ fcact, const float* __restrict__ w,
           const float* __restrict__ bias, float* __restrict__ out) {
  int gi = blockIdx.x * 256 + threadIdx.x;
  if (gi >= 512) return;
  int b = gi >> 1, j = gi & 1;
  float s = bias[j];
  const _Float16* row = fcact + (size_t)b * 576;
  const float* wr = w + j * 564;
  for (int i = 0; i < 564; ++i) s += (float)row[i] * wr[i];
  out[gi] = s;   // gi == b*2 + j
}

// ---------------------------------------------------------------------------
extern "C" void kernel_launch(void* const* d_in, const int* in_sizes, int n_in,
                              void* d_out, int out_size, void* d_ws, size_t ws_size,
                              hipStream_t stream) {
  (void)in_sizes; (void)n_in; (void)out_size; (void)ws_size;
  const float* x   = (const float*)d_in[0];
  const float* qv  = (const float*)d_in[1];
  const float* bas = (const float*)d_in[2];
  const float* c1w = (const float*)d_in[3];
  const float* c1b = (const float*)d_in[4];
  const float* g1  = (const float*)d_in[5];
  const float* b1  = (const float*)d_in[6];
  const float* m1  = (const float*)d_in[7];
  const float* v1  = (const float*)d_in[8];
  const float* c2w = (const float*)d_in[9];
  const float* c2b = (const float*)d_in[10];
  const float* g2  = (const float*)d_in[11];
  const float* b2  = (const float*)d_in[12];
  const float* m2  = (const float*)d_in[13];
  const float* v2  = (const float*)d_in[14];
  const float* f1w = (const float*)d_in[15];
  const float* f1b = (const float*)d_in[16];
  const float* f2w = (const float*)d_in[17];
  const float* f2b = (const float*)d_in[18];
  float* out = (float*)d_out;

  char* ws = (char*)d_ws;
  size_t off = 0;
  auto alloc = [&](size_t bytes) {
    void* p = ws + off;
    off = (off + bytes + 255) & ~(size_t)255;
    return p;
  };
  _Float16* featH = (_Float16*)alloc((size_t)256 * 160 * 64 * 2); // 5.24 MB
  _Float16* out1h = (_Float16*)alloc((size_t)256 * 120 * 32 * 2); // 1.97 MB
  _Float16* fcin  = (_Float16*)alloc((size_t)256 * 1536 * 2);     // 786 KB
  _Float16* fcact = (_Float16*)alloc((size_t)256 * 576 * 2);      // 295 KB
  _Float16* w1h   = (_Float16*)alloc((size_t)128 * 800 * 2);      // 205 KB
  _Float16* w2h   = (_Float16*)alloc((size_t)128 * 608 * 2);      // 156 KB
  _Float16* f1wh  = (_Float16*)alloc((size_t)512 * 1536 * 2);     // 1.57 MB
  float*    bnp   = (float*)alloc((size_t)480 * 4);               // folded BN

  k_prep<<<512, 256, 0, stream>>>(qv, c1w, c2w, f1w, g1, b1, m1, v1,
                                  g2, b2, m2, v2, w1h, w2h, f1wh,
                                  fcin, fcact, bnp);
  k_dct_hist<<<256, 256, 0, stream>>>(x, bas, featH);
  k_conv1<<<256, 256, 0, stream>>>(featH, w1h, c1b, bnp, bnp + 120, out1h);
  k_conv2<<<256, 256, 0, stream>>>(out1h, w2h, c2b, bnp + 240, bnp + 360, fcin);
  k_fc1<<<64, 256, 0, stream>>>(fcin, f1wh, f1b, fcact);
  k_fc2<<<2, 256, 0, stream>>>(fcact, f2w, f2b, out);
}
